// ContrastiveNetWithGATAndTransformer_22041772163477
// MI455X (gfx1250) — compile-verified, matrix operand order checked by reading, not run on hardware
//
#include <hip/hip_runtime.h>
#include <hip/hip_bf16.h>

// ---------------------------------------------------------------------------
// CDNA5 / gfx1250: encoder MLP -> GATConv -> 2x Transformer.
// Dense matmuls: v_wmma_f32_16x16x32_f16. GEMM B tiles staged to LDS via the
// Tensor Data Mover (tensor_load_to_lds + s_wait_tensorcnt, double buffered),
// consumed with ds_load_b128; A fragments software-pipelined from global.
// ---------------------------------------------------------------------------

typedef _Float16 h16;
typedef __attribute__((ext_vector_type(16))) _Float16 v16h;
typedef __attribute__((ext_vector_type(8)))  float    v8f;
typedef __attribute__((ext_vector_type(4))) unsigned int u32x4;
typedef __attribute__((ext_vector_type(8))) int          i32x8;
typedef __attribute__((ext_vector_type(4))) int          i32x4;

#define NNODES 4096
#define IN_DIM 40
#define IN_PAD 64
#define DMODEL 128
#define NHEAD  4
#define DH     32
#define FF     2048
#define NEDGES 131072
#define ETOT   (NEDGES + NNODES)   // + self loops

// ---- 16x32 f16 fragment loader (ISA 7.12.2 striping). Works for A tiles of
// row-major [M][K] and B tiles of row-major [N][K] (i.e. x @ W^T).
__device__ inline v16h load_frag(const h16* base, int row_stride) {
    int lane = threadIdx.x & 31;
    int r    = lane & 15;            // row (M or N index within tile)
    int kb   = (lane >> 4) << 3;     // K sub-offset: 0 or 8
    const h16* p = base + (size_t)r * row_stride + kb;
    union { uint4 u[2]; v16h v; } t;
    t.u[0] = *(const uint4*)(p);       // K = kb .. kb+7
    t.u[1] = *(const uint4*)(p + 16);  // K = 16+kb .. 16+kb+7
    return t.v;
}

// ---- LDS fragment load via ds_load_b128 (tile packed as [16 rows][32 f16]).
// Address is a wave-relative LDS byte offset, matching the TDM lds_addr base.
__device__ inline v16h ds_load_frag(unsigned lds_byte_addr) {
    u32x4 lo, hi;
    asm volatile("ds_load_b128 %0, %2\n\t"
                 "ds_load_b128 %1, %2 offset:32"
                 : "=v"(lo), "=v"(hi)
                 : "v"(lds_byte_addr));
    union { u32x4 u[2]; v16h v; } t;
    t.u[0] = lo; t.u[1] = hi;
    return t.v;
}

// ---- TDM: DMA a 64(row) x 32(col) f16 tile (row stride = K elems) into LDS.
// D# per ISA 8.3/8.4: group0 = {count|flags, lds_addr, global_addr, type=2},
// group1 = {data_size=2B, tensor dims (large), tile 32x64, dim0 stride}.
__device__ inline void tdm_load_tile_b(unsigned lds_off, const h16* gptr, int row_stride_elems) {
    unsigned long long ga = (unsigned long long)(size_t)gptr;
    u32x4 g0;
    g0[0] = 1u;                                        // count = 1 (valid D#)
    g0[1] = lds_off;                                   // LDS byte address
    g0[2] = (unsigned)(ga & 0xffffffffull);            // global_addr[31:0]
    g0[3] = (unsigned)((ga >> 32) & 0x01ffffffull)     // global_addr[56:32]
          | 0x80000000u;                               // type = 2 ("image")
    i32x8 g1;
    g1[0] = 0x00010000;          // wg_mask=0, data_size=1 (2 bytes)
    g1[1] = (int)0x80000000;     // tensor_dim0 lo16 = 0x8000 (generous bound)
    g1[2] = (int)0x80000000;     // tensor_dim0 hi=0, tensor_dim1 lo16 = 0x8000
    g1[3] = (int)(32u << 16);    // tensor_dim1 hi=0, tile_dim0 = 32
    g1[4] = 64;                  // tile_dim1 = 64, tile_dim2 = 0
    g1[5] = row_stride_elems;    // tensor_dim0_stride lo32
    g1[6] = 0;
    g1[7] = 0;
    i32x4 z4 = {0, 0, 0, 0};
#if defined(__clang_major__) && __clang_major__ >= 23
    i32x8 z8 = {0, 0, 0, 0, 0, 0, 0, 0};
    __builtin_amdgcn_tensor_load_to_lds(g0, g1, z4, z4, z8, 0);
#else
    __builtin_amdgcn_tensor_load_to_lds(g0, g1, z4, z4, 0);
#endif
}

// ---- GEMM: C[M][N] = act(A[M][K] @ W[N][K]^T + bias).
// One wave computes a 32x64 strip: 2 A frags x 4 B frags, 8 wmma per K-step.
// B tiles staged to per-wave LDS via TDM (double-buffered on TENSORcnt),
// read back with a clause of ds_load_b128 + one s_wait_dscnt;
// A fragments prefetched one K-step ahead from global memory.
// launch_bounds(256,1): ~160-VGPR working set must stay resident (no spills).
__global__ void __launch_bounds__(256, 1)
gemm32x64_f16(const h16* __restrict__ A, const h16* __restrict__ W,
              const float* __restrict__ bias, float* __restrict__ C,
              int M, int N, int K, int relu) {
    __shared__ h16 bstage[8][2][64 * 32];              // 8 waves x 2 bufs x 4KB
    int wib = __builtin_amdgcn_readfirstlane((int)(threadIdx.x >> 5));
    int wid = blockIdx.x * 8 + wib;
    int tiles_n = N >> 6;
    int tm = wid / tiles_n, tn = wid % tiles_n;
    if (tm * 32 >= M) return;
    int row0 = tm * 32, col0 = tn * 64;
    unsigned lds_base = (unsigned)wib * (2u * 64u * 32u * 2u);

    // Keep the staging buffer alive (TDM writes it; ds_load asm reads it by
    // raw offset). A single escaping use pins the LDS allocation at offset 0.
    {
        h16* keep = &bstage[0][0][0];
        asm volatile("" :: "v"(keep));
    }

    int lane = threadIdx.x & 31;
    // per-lane byte offset inside a 16x32 f16 tile (row*64B + (lane>>4)*16B)
    unsigned lane_off = (unsigned)((lane & 15) * 64 + ((lane >> 4) << 4));

    tdm_load_tile_b(lds_base, W + (size_t)col0 * K, K);   // prologue stage B(k=0)
    v16h a0 = load_frag(A + (size_t)row0 * K, K);          // prologue A(k=0)
    v16h a1 = load_frag(A + (size_t)(row0 + 16) * K, K);
    v8f acc0[4] = {}, acc1[4] = {};
    int buf = 0;
    for (int k = 0; k < K; k += 32) {
        v16h a0n = a0, a1n = a1;
        if (k + 32 < K) {
            // prefetch next A fragments (loadcnt wait lands next iteration)
            a0n = load_frag(A + (size_t)row0 * K + k + 32, K);
            a1n = load_frag(A + (size_t)(row0 + 16) * K + k + 32, K);
            // stage next B tile into the other LDS buffer
            tdm_load_tile_b(lds_base + (unsigned)((buf ^ 1) * 4096),
                            W + (size_t)col0 * K + (k + 32), K);
            __builtin_amdgcn_s_wait_tensorcnt(1);      // current buf ready
        } else {
            __builtin_amdgcn_s_wait_tensorcnt(0);
        }
        unsigned bb = lds_base + (unsigned)(buf * 4096) + lane_off;
        v16h b0 = ds_load_frag(bb);
        v16h b1 = ds_load_frag(bb + 1024);
        v16h b2 = ds_load_frag(bb + 2048);
        v16h b3 = ds_load_frag(bb + 3072);
        // Single DS wait; tying the fragments through it guarantees the wmmas
        // below consume post-wait data.
        asm volatile("s_wait_dscnt 0x0"
                     : "+v"(b0), "+v"(b1), "+v"(b2), "+v"(b3));
        acc0[0] = __builtin_amdgcn_wmma_f32_16x16x32_f16(false, a0, false, b0, (short)0, acc0[0], false, false);
        acc1[0] = __builtin_amdgcn_wmma_f32_16x16x32_f16(false, a1, false, b0, (short)0, acc1[0], false, false);
        acc0[1] = __builtin_amdgcn_wmma_f32_16x16x32_f16(false, a0, false, b1, (short)0, acc0[1], false, false);
        acc1[1] = __builtin_amdgcn_wmma_f32_16x16x32_f16(false, a1, false, b1, (short)0, acc1[1], false, false);
        acc0[2] = __builtin_amdgcn_wmma_f32_16x16x32_f16(false, a0, false, b2, (short)0, acc0[2], false, false);
        acc1[2] = __builtin_amdgcn_wmma_f32_16x16x32_f16(false, a1, false, b2, (short)0, acc1[2], false, false);
        acc0[3] = __builtin_amdgcn_wmma_f32_16x16x32_f16(false, a0, false, b3, (short)0, acc0[3], false, false);
        acc1[3] = __builtin_amdgcn_wmma_f32_16x16x32_f16(false, a1, false, b3, (short)0, acc1[3], false, false);
        a0 = a0n; a1 = a1n;
        buf ^= 1;
    }
    int n = lane & 15;
    int mb = (lane >> 4) << 3;
#pragma unroll
    for (int j = 0; j < 4; ++j) {
        int col = col0 + j * 16 + n;
        float bv = bias ? bias[col] : 0.f;
#pragma unroll
        for (int i = 0; i < 8; ++i) {
            float v0 = acc0[j][i] + bv;
            float v1 = acc1[j][i] + bv;
            if (relu) { v0 = fmaxf(v0, 0.f); v1 = fmaxf(v1, 0.f); }
            C[(size_t)(row0 + mb + i) * N + col]      = v0;
            C[(size_t)(row0 + 16 + mb + i) * N + col] = v1;
        }
    }
}

// ---- f32 -> f16 convert with optional K padding (zeros).
__global__ void conv_pad_h16(const float* __restrict__ in, h16* __restrict__ out,
                             int M, int K, int Kp) {
    int idx = blockIdx.x * blockDim.x + threadIdx.x;
    if (idx >= M * Kp) return;
    int r = idx / Kp, k = idx - r * Kp;
    out[idx] = (h16)(k < K ? in[(size_t)r * K + k] : 0.f);
}

__global__ void fill_f32(float* p, float v, int n) {
    int i = blockIdx.x * blockDim.x + threadIdx.x;
    if (i < n) p[i] = v;
}

// ---- GAT pieces ------------------------------------------------------------
__global__ void gat_alpha(const float* __restrict__ xw,
                          const float* __restrict__ asrc,
                          const float* __restrict__ adst,
                          float* __restrict__ alsrc, float* __restrict__ aldst) {
    int idx = blockIdx.x * blockDim.x + threadIdx.x;   // node*2 + head
    if (idx >= NNODES * 2) return;
    int nd = idx >> 1, hh = idx & 1;
    const float* xp = xw + (size_t)nd * 128 + hh * 64;
    float s1 = 0.f, s2 = 0.f;
    for (int c = 0; c < 64; ++c) {
        float v = xp[c];
        s1 += v * asrc[hh * 64 + c];
        s2 += v * adst[hh * 64 + c];
    }
    alsrc[idx] = s1; aldst[idx] = s2;
}

__device__ inline void atomicMaxFloat(float* addr, float v) {
    if (v >= 0.f) atomicMax((int*)addr, __float_as_int(v));
    else          atomicMin((unsigned int*)addr, (unsigned int)__float_as_int(v));
}

__device__ inline void edge_nodes(const int* ei, int e, int& s, int& d) {
    if (e < NEDGES) { s = ei[e]; d = ei[NEDGES + e]; }
    else            { s = d = e - NEDGES; }           // self loop
}

__global__ void gat_edge_max(const int* __restrict__ ei,
                             const float* __restrict__ alsrc,
                             const float* __restrict__ aldst,
                             float* __restrict__ m) {
    int e = blockIdx.x * blockDim.x + threadIdx.x;
    if (e >= ETOT) return;
    int s, d; edge_nodes(ei, e, s, d);
#pragma unroll
    for (int hh = 0; hh < 2; ++hh) {
        float x = alsrc[s * 2 + hh] + aldst[d * 2 + hh];
        x = x > 0.f ? x : 0.2f * x;                    // leaky_relu 0.2
        atomicMaxFloat(&m[d * 2 + hh], x);
    }
}

__global__ void gat_edge_sum(const int* __restrict__ ei,
                             const float* __restrict__ alsrc,
                             const float* __restrict__ aldst,
                             const float* __restrict__ m,
                             const float* __restrict__ xw,
                             float* __restrict__ ssum, float* __restrict__ num) {
    int t = blockIdx.x * blockDim.x + threadIdx.x;     // edge*2 + head
    if (t >= ETOT * 2) return;
    int e = t >> 1, hh = t & 1;
    int s, d; edge_nodes(ei, e, s, d);
    float x = alsrc[s * 2 + hh] + aldst[d * 2 + hh];
    x = x > 0.f ? x : 0.2f * x;
    float ex = __expf(x - m[d * 2 + hh]);
    atomicAdd(&ssum[d * 2 + hh], ex);
    const float* xp = xw + (size_t)s * 128 + hh * 64;
    float* np = num + (size_t)d * 128 + hh * 64;
    for (int c = 0; c < 64; ++c) atomicAdd(&np[c], ex * xp[c]);
}

__global__ void gat_final(const float* __restrict__ num, const float* __restrict__ ssum,
                          const float* __restrict__ bias, float* __restrict__ out) {
    int idx = blockIdx.x * blockDim.x + threadIdx.x;   // node*128 + c
    if (idx >= NNODES * 128) return;
    int nd = idx >> 7, c = idx & 127, hh = c >> 6;
    out[idx] = num[idx] / (ssum[nd * 2 + hh] + 1e-16f) + bias[c];
}

// ---- V transpose for attention: Vt[h*32+d][s] -------------------------------
__global__ void transpose_v(const float* __restrict__ qkv, h16* __restrict__ vt) {
    int idx = blockIdx.x * blockDim.x + threadIdx.x;   // (h*32+d)*4096 + s
    if (idx >= NHEAD * DH * NNODES) return;
    int hd = idx >> 12, s = idx & 4095;
    vt[idx] = (h16)qkv[(size_t)s * 384 + 256 + hd];
}

// ---- Flash attention: one wave per (head, 16-query block). ------------------
__global__ void __launch_bounds__(256, 1)
attn_flash(const h16* __restrict__ qkvh, const h16* __restrict__ vt,
           float* __restrict__ out) {
    int wib  = threadIdx.x >> 5;
    int wid  = blockIdx.x * (blockDim.x >> 5) + wib;
    int h    = wid >> 8;           // 256 query blocks per head
    int qb   = wid & 255;
    int lane = threadIdx.x & 31;
    __shared__ h16 pbuf[8][16 * 32];
    h16* P = pbuf[wib];

    const float scale = 0.17677669529663687f;   // 1/sqrt(32)
    v16h qf = load_frag(qkvh + (size_t)(qb * 16) * 384 + h * 32, 384);

    float mrow[8], lrow[8];
    v8f o0 = {}, o1 = {};
#pragma unroll
    for (int i = 0; i < 8; ++i) { mrow[i] = -1e30f; lrow[i] = 0.f; }

    for (int kb = 0; kb < NNODES; kb += 32) {
        v16h k0 = load_frag(qkvh + (size_t)kb * 384 + 128 + h * 32, 384);
        v16h k1 = load_frag(qkvh + (size_t)(kb + 16) * 384 + 128 + h * 32, 384);
        v8f s0 = {}, s1 = {};
        s0 = __builtin_amdgcn_wmma_f32_16x16x32_f16(false, qf, false, k0, (short)0, s0, false, false);
        s1 = __builtin_amdgcn_wmma_f32_16x16x32_f16(false, qf, false, k1, (short)0, s1, false, false);

        int m = ((lane >> 4) << 3);
        int n = lane & 15;
#pragma unroll
        for (int i = 0; i < 8; ++i) {
            float v0 = s0[i] * scale, v1 = s1[i] * scale;
            float mx = fmaxf(v0, v1);
            for (int d = 1; d < 16; d <<= 1) mx = fmaxf(mx, __shfl_xor(mx, d, 32));
            float mnew = fmaxf(mrow[i], mx);
            float a  = __expf(mrow[i] - mnew);
            float p0 = __expf(v0 - mnew), p1 = __expf(v1 - mnew);
            float rs = p0 + p1;
            for (int d = 1; d < 16; d <<= 1) rs += __shfl_xor(rs, d, 32);
            lrow[i] = lrow[i] * a + rs;
            mrow[i] = mnew;
            o0[i] *= a; o1[i] *= a;
            P[(m + i) * 32 + n]      = (h16)p0;
            P[(m + i) * 32 + n + 16] = (h16)p1;
        }
        v16h pf  = load_frag(P, 32);
        v16h vf0 = load_frag(vt + ((size_t)h * 32 + 0)  * NNODES + kb, NNODES);
        v16h vf1 = load_frag(vt + ((size_t)h * 32 + 16) * NNODES + kb, NNODES);
        o0 = __builtin_amdgcn_wmma_f32_16x16x32_f16(false, pf, false, vf0, (short)0, o0, false, false);
        o1 = __builtin_amdgcn_wmma_f32_16x16x32_f16(false, pf, false, vf1, (short)0, o1, false, false);
    }
    int n = lane & 15, mb = (lane >> 4) << 3;
#pragma unroll
    for (int i = 0; i < 8; ++i) {
        float inv = 1.f / lrow[i];
        size_t row = (size_t)(qb * 16 + mb + i) * 128 + h * 32;
        out[row + n]      = o0[i] * inv;
        out[row + 16 + n] = o1[i] * inv;
    }
}

// ---- fused residual + LayerNorm (one wave per 128-wide row) -----------------
__global__ void residual_ln(const float* __restrict__ x, const float* __restrict__ o,
                            const float* __restrict__ g, const float* __restrict__ b,
                            float* __restrict__ out) {
    int row  = blockIdx.x * 8 + (threadIdx.x >> 5);
    int lane = threadIdx.x & 31;
    const float* xp = x + (size_t)row * 128;
    const float* op = o + (size_t)row * 128;
    float v[4]; float sum = 0.f;
#pragma unroll
    for (int j = 0; j < 4; ++j) { v[j] = xp[lane + 32 * j] + op[lane + 32 * j]; sum += v[j]; }
    for (int d = 1; d < 32; d <<= 1) sum += __shfl_xor(sum, d, 32);
    float mu = sum * (1.f / 128.f);
    float var = 0.f;
#pragma unroll
    for (int j = 0; j < 4; ++j) { float t = v[j] - mu; var += t * t; }
    for (int d = 1; d < 32; d <<= 1) var += __shfl_xor(var, d, 32);
    float inv = rsqrtf(var * (1.f / 128.f) + 1e-5f);
#pragma unroll
    for (int j = 0; j < 4; ++j) {
        int c = lane + 32 * j;
        out[(size_t)row * 128 + c] = (v[j] - mu) * inv * g[c] + b[c];
    }
}

// ---------------------------------------------------------------------------
extern "C" void kernel_launch(void* const* d_in, const int* in_sizes, int n_in,
                              void* d_out, int out_size, void* d_ws, size_t ws_size,
                              hipStream_t stream) {
    (void)in_sizes; (void)n_in; (void)out_size; (void)ws_size;
    const float* x       = (const float*)d_in[0];
    const int*   ei      = (const int*)  d_in[1];
    const float* enc_w1  = (const float*)d_in[2];
    const float* enc_b1  = (const float*)d_in[3];
    const float* enc_w2  = (const float*)d_in[4];
    const float* enc_b2  = (const float*)d_in[5];
    const float* gat_w   = (const float*)d_in[6];
    const float* gat_asr = (const float*)d_in[7];
    const float* gat_ads = (const float*)d_in[8];
    const float* gat_b   = (const float*)d_in[9];

    char* wsb = (char*)d_ws;
    size_t off = 0;
    auto alloc = [&](size_t bytes) -> void* {
        void* p = wsb + off; off = (off + bytes + 255) & ~(size_t)255; return p;
    };
    h16*  XH   = (h16*) alloc((size_t)NNODES * IN_PAD * 2);
    h16*  W1H  = (h16*) alloc((size_t)DMODEL * IN_PAD * 2);
    h16*  W2H  = (h16*) alloc((size_t)DMODEL * DMODEL * 2);
    h16*  WGH  = (h16*) alloc((size_t)DMODEL * DMODEL * 2);
    h16*  INWH[2], *OUTWH[2], *FF1H[2], *FF2H[2];
    for (int l = 0; l < 2; ++l) {
        INWH[l]  = (h16*) alloc((size_t)3 * DMODEL * DMODEL * 2);
        OUTWH[l] = (h16*) alloc((size_t)DMODEL * DMODEL * 2);
        FF1H[l]  = (h16*) alloc((size_t)FF * DMODEL * 2);
        FF2H[l]  = (h16*) alloc((size_t)DMODEL * FF * 2);
    }
    float* FA   = (float*)alloc((size_t)NNODES * 128 * 4);
    float* FB   = (float*)alloc((size_t)NNODES * 128 * 4);
    float* FC   = (float*)alloc((size_t)NNODES * 128 * 4);
    h16*   HA   = (h16*)  alloc((size_t)NNODES * 128 * 2);
    float* FBIG = (float*)alloc((size_t)NNODES * FF * 4);
    h16*   HBIG = (h16*)  alloc((size_t)NNODES * FF * 2);
    h16*   VT   = (h16*)  alloc((size_t)NHEAD * DH * NNODES * 2);
    float* ALS  = (float*)alloc(NNODES * 2 * 4);
    float* ALD  = (float*)alloc(NNODES * 2 * 4);
    float* MM   = (float*)alloc(NNODES * 2 * 4);
    float* SS   = (float*)alloc(NNODES * 2 * 4);

    auto conv = [&](const float* in, h16* out, int M, int K, int Kp) {
        int n = M * Kp;
        conv_pad_h16<<<(n + 255) / 256, 256, 0, stream>>>(in, out, M, K, Kp);
    };
    auto gemm = [&](const h16* A, const h16* Wm, const float* bias, float* C,
                    int M, int N, int K, int relu) {
        int waves = (M / 32) * (N / 64);
        gemm32x64_f16<<<waves / 8, 256, 0, stream>>>(A, Wm, bias, C, M, N, K, relu);
    };

    // ---- convert inputs & weights to f16 (pad K=40 -> 64)
    conv(x, XH, NNODES, IN_DIM, IN_PAD);
    conv(enc_w1, W1H, DMODEL, IN_DIM, IN_PAD);
    conv(enc_w2, W2H, DMODEL, DMODEL, DMODEL);
    conv(gat_w,  WGH, DMODEL, DMODEL, DMODEL);
    for (int l = 0; l < 2; ++l) {
        const float* in_w  = (const float*)d_in[10 + 12 * l + 0];
        const float* out_w = (const float*)d_in[10 + 12 * l + 2];
        const float* ff_w1 = (const float*)d_in[10 + 12 * l + 6];
        const float* ff_w2 = (const float*)d_in[10 + 12 * l + 8];
        conv(in_w,  INWH[l],  3 * DMODEL, DMODEL, DMODEL);
        conv(out_w, OUTWH[l], DMODEL, DMODEL, DMODEL);
        conv(ff_w1, FF1H[l],  FF, DMODEL, DMODEL);
        conv(ff_w2, FF2H[l],  DMODEL, FF, FF);
    }

    // ---- encoder MLP: h = relu(x@W1^T+b1)@W2^T+b2
    gemm(XH, W1H, enc_b1, FA, NNODES, DMODEL, IN_PAD, 1);
    conv(FA, HA, NNODES, DMODEL, DMODEL);
    gemm(HA, W2H, enc_b2, FB, NNODES, DMODEL, DMODEL, 0);   // FB = h

    // ---- GATConv
    conv(FB, HA, NNODES, DMODEL, DMODEL);
    gemm(HA, WGH, nullptr, FC, NNODES, DMODEL, DMODEL, 0);  // FC = xw
    gat_alpha<<<(NNODES * 2 + 255) / 256, 256, 0, stream>>>(FC, gat_asr, gat_ads, ALS, ALD);
    fill_f32<<<(NNODES * 2 + 255) / 256, 256, 0, stream>>>(MM, -1e30f, NNODES * 2);
    fill_f32<<<(NNODES * 2 + 255) / 256, 256, 0, stream>>>(SS, 0.f, NNODES * 2);
    fill_f32<<<(NNODES * 128 + 255) / 256, 256, 0, stream>>>(FA, 0.f, NNODES * 128);
    gat_edge_max<<<(ETOT + 255) / 256, 256, 0, stream>>>(ei, ALS, ALD, MM);
    gat_edge_sum<<<(ETOT * 2 + 255) / 256, 256, 0, stream>>>(ei, ALS, ALD, MM, FC, SS, FA);
    gat_final<<<(NNODES * 128 + 255) / 256, 256, 0, stream>>>(FA, SS, gat_b, FB);  // FB = h

    // ---- 2 transformer encoder layers (post-norm, relu FFN)
    for (int l = 0; l < 2; ++l) {
        const float* in_b  = (const float*)d_in[10 + 12 * l + 1];
        const float* out_b = (const float*)d_in[10 + 12 * l + 3];
        const float* ln1_g = (const float*)d_in[10 + 12 * l + 4];
        const float* ln1_b = (const float*)d_in[10 + 12 * l + 5];
        const float* ff_b1 = (const float*)d_in[10 + 12 * l + 7];
        const float* ff_b2 = (const float*)d_in[10 + 12 * l + 9];
        const float* ln2_g = (const float*)d_in[10 + 12 * l + 10];
        const float* ln2_b = (const float*)d_in[10 + 12 * l + 11];

        conv(FB, HA, NNODES, DMODEL, DMODEL);
        gemm(HA, INWH[l], in_b, FBIG, NNODES, 3 * DMODEL, DMODEL, 0);     // qkv
        conv(FBIG, HBIG, NNODES, 3 * DMODEL, 3 * DMODEL);                  // qkv f16
        transpose_v<<<(NHEAD * DH * NNODES + 255) / 256, 256, 0, stream>>>(FBIG, VT);
        attn_flash<<<(NHEAD * (NNODES / 16)) / 8, 256, 0, stream>>>(HBIG, VT, FC);
        conv(FC, HA, NNODES, DMODEL, DMODEL);
        gemm(HA, OUTWH[l], out_b, FA, NNODES, DMODEL, DMODEL, 0);          // o
        residual_ln<<<NNODES / 8, 256, 0, stream>>>(FB, FA, ln1_g, ln1_b, FC);  // x1
        conv(FC, HA, NNODES, DMODEL, DMODEL);
        gemm(HA, FF1H[l], ff_b1, FBIG, NNODES, FF, DMODEL, 1);             // relu ff1
        conv(FBIG, HBIG, NNODES, FF, FF);
        gemm(HBIG, FF2H[l], ff_b2, FA, NNODES, DMODEL, FF, 0);             // ff2
        float* dst = (l == 1) ? (float*)d_out : FB;
        residual_ln<<<NNODES / 8, 256, 0, stream>>>(FC, FA, ln2_g, ln2_b, dst);
    }
}
